// SpatioTemporalPINN_90383291777515
// MI455X (gfx1250) — matrix operand order repaired and checked by reading, not compiled, not run on hardware
//
#include <hip/hip_runtime.h>
#include <math.h>

typedef float  v8f   __attribute__((ext_vector_type(8)));
typedef __bf16 v8bf  __attribute__((ext_vector_type(8)));
typedef __bf16 v16bf __attribute__((ext_vector_type(16)));

#define NN 512   // nodes
#define CC 32    // residual channels
#define BB 32    // batch
#define TT 64    // initial time length
#define YOFF 196608  // 32*512*12, offset of res part in d_out

// ---------------------------------------------------------------------------
// adj = softmax(relu(adp), axis=1) + pre ; stored transposed bf16: adjT[m][n]
// ---------------------------------------------------------------------------
__global__ __launch_bounds__(256) void adj_prep_kernel(
    const float* __restrict__ adp, const float* __restrict__ pre,
    __bf16* __restrict__ adjT) {
  __shared__ float red[256];
  int row = blockIdx.x, tid = threadIdx.x;
  float mx = 0.f;                       // relu output >= 0
  for (int m = tid; m < NN; m += 256) {
    float v = adp[row * NN + m]; v = v > 0.f ? v : 0.f;
    mx = fmaxf(mx, v);
  }
  red[tid] = mx; __syncthreads();
  for (int s = 128; s; s >>= 1) { if (tid < s) red[tid] = fmaxf(red[tid], red[tid + s]); __syncthreads(); }
  mx = red[0]; __syncthreads();
  float sum = 0.f;
  for (int m = tid; m < NN; m += 256) {
    float v = adp[row * NN + m]; v = v > 0.f ? v : 0.f;
    sum += expf(v - mx);
  }
  red[tid] = sum; __syncthreads();
  for (int s = 128; s; s >>= 1) { if (tid < s) red[tid] += red[tid + s]; __syncthreads(); }
  float inv = 1.f / red[0];
  for (int m = tid; m < NN; m += 256) {
    float v = adp[row * NN + m]; v = v > 0.f ? v : 0.f;
    float a = expf(v - mx) * inv + pre[row * NN + m];
    adjT[(size_t)m * NN + row] = (__bf16)a;
  }
}

// ---------------------------------------------------------------------------
// x0[b,t,c,n] = start_w[c] * x_data[b,0,n,t] + start_b[c]
// ---------------------------------------------------------------------------
__global__ __launch_bounds__(256) void start_conv_kernel(
    const float* __restrict__ xd, const float* __restrict__ sw,
    const float* __restrict__ sb, float* __restrict__ X) {
  int idx = blockIdx.x * 256 + threadIdx.x;   // E = 32<<20
  int n = idx & (NN - 1);
  int c = (idx >> 9) & 31;
  int t = (idx >> 14) & 63;
  int b = idx >> 20;
  float x0 = xd[((size_t)b * NN + n) * TT + t];
  X[idx] = sw[c] * x0 + sb[c];                // layout matches decomposition
}

// ---------------------------------------------------------------------------
// gated dilated conv: xc = tanh(W_f*x) * sigmoid(W_g*x), bf16 out (+f32 last t)
// ---------------------------------------------------------------------------
__global__ __launch_bounds__(256) void dilconv_kernel(
    const float* __restrict__ X, const float* __restrict__ fw,
    const float* __restrict__ fb, const float* __restrict__ gw,
    const float* __restrict__ gb, __bf16* __restrict__ xc,
    float* __restrict__ xclast, int T_in, int T_out, int d) {
  int idx = blockIdx.x * 256 + threadIdx.x;
  int n = idx & 511;
  int c = (idx >> 9) & 31;
  int rest = idx >> 14;
  int t = rest % T_out;
  int b = rest / T_out;
  if (b >= BB) return;
  const float* x0 = X + ((size_t)(b * T_in + t) * CC) * NN + n;
  const float* x1 = X + ((size_t)(b * T_in + t + d) * CC) * NN + n;
  float f = fb[c], g = gb[c];
#pragma unroll
  for (int i = 0; i < 32; i++) {
    float a0 = x0[i * NN], a1 = x1[i * NN];
    f += fw[c * 64 + i * 2] * a0 + fw[c * 64 + i * 2 + 1] * a1;
    g += gw[c * 64 + i * 2] * a0 + gw[c * 64 + i * 2 + 1] * a1;
  }
  float v = tanhf(f) * (1.f / (1.f + expf(-g)));
  xc[((size_t)(b * T_out + t) * CC + c) * NN + n] = (__bf16)v;
  if (t == T_out - 1) xclast[((size_t)b * CC + c) * NN + n] = v;
}

// ---------------------------------------------------------------------------
// skip contribution (only last time index feeds the heads — provably enough)
// ---------------------------------------------------------------------------
__global__ __launch_bounds__(256) void skip_kernel(
    const float* __restrict__ xclast, const float* __restrict__ sw,
    const float* __restrict__ sb, float* __restrict__ skip, int init) {
  int idx = blockIdx.x * 256 + threadIdx.x;   // E = 32*256*512
  int n = idx & 511;
  int o = (idx >> 9) & 255;
  int b = idx >> 17;
  float acc = sb[o];
#pragma unroll
  for (int ci = 0; ci < 32; ci++)
    acc += sw[o * 32 + ci] * xclast[((size_t)b * 32 + ci) * NN + n];
  if (init) skip[idx] = acc; else skip[idx] += acc;
}

// ---------------------------------------------------------------------------
// fused: xg = x_conv @ adj (bf16 WMMA, f32 acc)  ->  x = rw@xg + rb + residual
// block: 128 thr (4 waves), tile M=32(c) x N=64(m), K=512(n)
// ---------------------------------------------------------------------------
__global__ __launch_bounds__(128) void graphres_kernel(
    const __bf16* __restrict__ xc, const __bf16* __restrict__ adjT,
    const float* __restrict__ rw, const float* __restrict__ rb,
    const float* __restrict__ xprev, float* __restrict__ xnext,
    int T_in, int T_out, int d) {
  __shared__ float xg[32 * 66];
  __shared__ float rwsh[1024];
  __shared__ float rbsh[32];
  int tid = threadIdx.x;
  int b = blockIdx.z, t = blockIdx.y;
  int n0 = blockIdx.x * 64;
#pragma unroll
  for (int q = 0; q < 8; q++) rwsh[tid + q * 128] = rw[tid + q * 128];
  if (tid < 32) rbsh[tid] = rb[tid];

  int lane = tid & 31, wv = tid >> 5;
  int mb = (wv >> 1) * 16, nb = (wv & 1) * 32;
  int half = lane >> 4;
  // A fragment (contiguous k order, same as B): lane<16 -> K {0..15}, lane>=16 -> K {16..31}
  const __bf16* aptr = xc + ((size_t)(b * T_out + t) * CC + (mb + (lane & 15))) * NN + half * 16;
  // B fragment: lane holds col m fixed, 16 contiguous K values from adjT[m][k..]
  const __bf16* bptr0 = adjT + (size_t)(n0 + nb + (lane & 15)) * NN + half * 16;
  const __bf16* bptr1 = bptr0 + (size_t)16 * NN;

  v8f acc0 = {0.f, 0.f, 0.f, 0.f, 0.f, 0.f, 0.f, 0.f};
  v8f acc1 = {0.f, 0.f, 0.f, 0.f, 0.f, 0.f, 0.f, 0.f};
  for (int kk = 0; kk < 16; kk++) {
    int k0 = kk * 32;
    v8bf alo = *(const v8bf*)(aptr + k0);
    v8bf ahi = *(const v8bf*)(aptr + k0 + 8);
    v16bf a = __builtin_shufflevector(alo, ahi, 0, 1, 2, 3, 4, 5, 6, 7, 8, 9, 10, 11, 12, 13, 14, 15);
    v8bf b0l = *(const v8bf*)(bptr0 + k0);
    v8bf b0h = *(const v8bf*)(bptr0 + k0 + 8);
    v16bf fb0 = __builtin_shufflevector(b0l, b0h, 0, 1, 2, 3, 4, 5, 6, 7, 8, 9, 10, 11, 12, 13, 14, 15);
    v8bf b1l = *(const v8bf*)(bptr1 + k0);
    v8bf b1h = *(const v8bf*)(bptr1 + k0 + 8);
    v16bf fb1 = __builtin_shufflevector(b1l, b1h, 0, 1, 2, 3, 4, 5, 6, 7, 8, 9, 10, 11, 12, 13, 14, 15);
    acc0 = __builtin_amdgcn_wmma_f32_16x16x32_bf16(false, a, false, fb0, (short)0, acc0, false, false);
    acc1 = __builtin_amdgcn_wmma_f32_16x16x32_bf16(false, a, false, fb1, (short)0, acc1, false, false);
    __builtin_prefetch(aptr + k0 + 128, 0, 1);   // -> global_prefetch_b8
    __builtin_prefetch(bptr0 + k0 + 128, 0, 1);
  }
  // D layout: vgpr r, lane l -> M = r + 8*(l>>4), N = l&15
#pragma unroll
  for (int r = 0; r < 8; r++) {
    int crow = mb + r + 8 * half;
    xg[crow * 66 + nb + (lane & 15)] = acc0[r];
    xg[crow * 66 + nb + 16 + (lane & 15)] = acc1[r];
  }
  __syncthreads();
  // epilogue: residual 1x1 conv + bias + residual add (trim = shift by d)
#pragma unroll
  for (int q = 0; q < 16; q++) {
    int flat = tid + q * 128;
    int oc = flat >> 6, on = flat & 63;
    float acc = rbsh[oc];
#pragma unroll
    for (int ci = 0; ci < 32; ci++) acc += rwsh[oc * 32 + ci] * xg[ci * 66 + on];
    acc += xprev[((size_t)(b * T_in + t + d) * CC + oc) * NN + n0 + on];
    xnext[((size_t)(b * T_out + t) * CC + oc) * NN + n0 + on] = acc;
  }
}

// ---------------------------------------------------------------------------
// feats_last = relu(e1w @ skip_last + e1b)
// ---------------------------------------------------------------------------
__global__ __launch_bounds__(256) void feats_kernel(
    const float* __restrict__ skip, const float* __restrict__ e1w,
    const float* __restrict__ e1b, float* __restrict__ feats) {
  int idx = blockIdx.x * 256 + threadIdx.x;   // E = 32*512*512
  int n = idx & 511;
  int o = (idx >> 9) & 511;
  int b = idx >> 18;
  float acc = e1b[o];
  for (int ci = 0; ci < 256; ci++)
    acc += e1w[o * 256 + ci] * skip[((size_t)b * 256 + ci) * NN + n];
  feats[idx] = fmaxf(acc, 0.f);
}

// ---------------------------------------------------------------------------
// heads: y_pred (B,N,12) into d_out, node features (B,N,32) into ws
// ---------------------------------------------------------------------------
__global__ __launch_bounds__(256) void heads_kernel(
    const float* __restrict__ feats, const float* __restrict__ e2w,
    const float* __restrict__ e2b, const float* __restrict__ fpw,
    const float* __restrict__ fpb, float* __restrict__ ypred,
    float* __restrict__ nf) {
  int idx = blockIdx.x * 256 + threadIdx.x;   // E = 32*512
  int n = idx & 511;
  int b = idx >> 9;
  float accy[12], accn[32];
#pragma unroll
  for (int o = 0; o < 12; o++) accy[o] = e2b[o];
#pragma unroll
  for (int c = 0; c < 32; c++) accn[c] = fpb[c];
  for (int ci = 0; ci < 512; ci++) {
    float fv = feats[((size_t)b * 512 + ci) * NN + n];
#pragma unroll
    for (int o = 0; o < 12; o++) accy[o] += e2w[o * 512 + ci] * fv;
#pragma unroll
    for (int c = 0; c < 32; c++) accn[c] += fpw[c * 512 + ci] * fv;
  }
#pragma unroll
  for (int o = 0; o < 12; o++) ypred[((size_t)b * 512 + n) * 12 + o] = accy[o];
#pragma unroll
  for (int c = 0; c < 32; c++) nf[((size_t)b * 512 + n) * 32 + c] = accn[c];
}

// ---------------------------------------------------------------------------
// PDE residual: one wave32 per point; exact 2nd-order directional jets through
// the RBF interpolation and the tanh MLP -> dT/dt, spatial Laplacian; q-net.
// ---------------------------------------------------------------------------
__global__ __launch_bounds__(256) void pde_kernel(
    const float* __restrict__ coords, const float* __restrict__ sensors,
    const float* __restrict__ nodefeat,
    const float* __restrict__ dw0, const float* __restrict__ db0,
    const float* __restrict__ dw1, const float* __restrict__ db1,
    const float* __restrict__ dw2, const float* __restrict__ db2,
    const float* __restrict__ dw3, const float* __restrict__ db3,
    const float* __restrict__ qw1, const float* __restrict__ qb1,
    const float* __restrict__ qg, const float* __restrict__ qbeta,
    const float* __restrict__ qw2, const float* __restrict__ qb2,
    const float* __restrict__ sigp, const float* __restrict__ alpp,
    float* __restrict__ out) {
  __shared__ float zv[8][2][64], zg[8][2][64], zh[8][2][64];
  __shared__ float ifsh[8][32];
  int lane = threadIdx.x & 31, wv = threadIdx.x >> 5;
  int point = blockIdx.x * 8 + wv;            // 4096 blocks * 8 waves = 32768
  int b = point >> 10;
  float c4a[4];
#pragma unroll
  for (int j = 0; j < 4; j++) c4a[j] = coords[(size_t)point * 4 + j];

  // distances to 512 sensors, 16 per lane
  float d2loc[16];
#pragma unroll
  for (int q = 0; q < 16; q++) {
    int sN = lane + q * 32;
    float dx = sensors[sN * 3 + 0] - c4a[0];
    float dy = sensors[sN * 3 + 1] - c4a[1];
    float dz = sensors[sN * 3 + 2] - c4a[2];
    d2loc[q] = dx * dx + dy * dy + dz * dz;
  }
  // 4 smallest (global) via per-lane min + butterfly reduce, exclude & repeat
  float td[4]; int ti[4];
  for (int k = 0; k < 4; k++) {
    float best = 3.4e38f; int bq = 0;
#pragma unroll
    for (int q = 0; q < 16; q++) if (d2loc[q] < best) { best = d2loc[q]; bq = q; }
    int bi = lane + bq * 32;
#pragma unroll
    for (int off = 16; off > 0; off >>= 1) {
      float od = __shfl_xor(best, off, 32);
      int   oi = __shfl_xor(bi, off, 32);
      if (od < best || (od == best && oi < bi)) { best = od; bi = oi; }
    }
    td[k] = best; ti[k] = bi;
    if ((bi & 31) == lane) d2loc[bi >> 5] = 3.4e38f;
  }
  float s = log1pf(expf(sigp[0])) + 0.01f;
  float inv2s2 = 1.f / (2.f * s * s);
  float wk[4], fk[4], S = 0.f;
#pragma unroll
  for (int k = 0; k < 4; k++) { wk[k] = expf(-td[k] * inv2s2); S += wk[k]; }
  float invD = 1.f / (S + 1e-5f);
#pragma unroll
  for (int k = 0; k < 4; k++)
    fk[k] = nodefeat[((size_t)(b * 512) + ti[k]) * 32 + lane];
  float ife = (wk[0] * fk[0] + wk[1] * fk[1] + wk[2] * fk[2] + wk[3] * fk[3]) * invD;
  ifsh[wv][lane] = ife;

  float lap = 0.f, dTdt = 0.f;
  for (int i = 0; i < 4; i++) {
    // jet of interpolated features along direction e_i (zero for time dir)
    float ig = 0.f, ihh = 0.f;
    if (i < 3) {
      float wp[4], wpp[4], Sp = 0.f, Spp = 0.f;
#pragma unroll
      for (int k = 0; k < 4; k++) {
        float pk = sensors[ti[k] * 3 + i];
        float dp = 2.f * (c4a[i] - pk);       // d(d2)/dxi ; d2'' = 2
        float t1 = dp * inv2s2;
        wp[k]  = -wk[k] * t1;
        wpp[k] =  wk[k] * (t1 * t1 - 2.f * inv2s2);
        Sp += wp[k]; Spp += wpp[k];
      }
#pragma unroll
      for (int k = 0; k < 4; k++) {
        float nwp  = wp[k] * invD - wk[k] * Sp * invD * invD;
        float nwpp = wpp[k] * invD - 2.f * wp[k] * Sp * invD * invD
                   - wk[k] * Spp * invD * invD
                   + 2.f * wk[k] * Sp * Sp * invD * invD * invD;
        ig += nwp * fk[k]; ihh += nwpp * fk[k];
      }
    }
    __syncthreads();
    // fill MLP input jet (36 entries): 4 coords + 32 interp features
    zv[wv][0][lane + 4] = ife; zg[wv][0][lane + 4] = ig; zh[wv][0][lane + 4] = ihh;
    if (lane < 4) {
      zv[wv][0][lane] = c4a[lane];
      zg[wv][0][lane] = (lane == i) ? 1.f : 0.f;
      zh[wv][0][lane] = 0.f;
    }
    __syncthreads();
    const float* Ws[3] = {dw0, dw1, dw2};
    const float* Bs[3] = {db0, db1, db2};
    const int dims[3] = {36, 64, 64};
    int pin = 0;
    for (int L = 0; L < 3; L++) {
      int nin = dims[L];
      const float* W = Ws[L]; const float* Bb = Bs[L];
      float sv0 = Bb[lane], sg0 = 0.f, sh0 = 0.f;
      float sv1 = Bb[lane + 32], sg1 = 0.f, sh1 = 0.f;
      for (int j = 0; j < nin; j++) {
        float vz = zv[wv][pin][j], gz = zg[wv][pin][j], hz = zh[wv][pin][j];
        float w0 = W[j * 64 + lane], w1 = W[j * 64 + lane + 32];
        sv0 += w0 * vz; sg0 += w0 * gz; sh0 += w0 * hz;
        sv1 += w1 * vz; sg1 += w1 * gz; sh1 += w1 * hz;
      }
      float tv0 = tanhf(sv0), m0 = 1.f - tv0 * tv0;
      float tg0 = m0 * sg0, th0 = m0 * sh0 - 2.f * tv0 * m0 * sg0 * sg0;
      float tv1 = tanhf(sv1), m1 = 1.f - tv1 * tv1;
      float tg1 = m1 * sg1, th1 = m1 * sh1 - 2.f * tv1 * m1 * sg1 * sg1;
      zv[wv][pin ^ 1][lane] = tv0; zg[wv][pin ^ 1][lane] = tg0; zh[wv][pin ^ 1][lane] = th0;
      zv[wv][pin ^ 1][lane + 32] = tv1; zg[wv][pin ^ 1][lane + 32] = tg1; zh[wv][pin ^ 1][lane + 32] = th1;
      __syncthreads();
      pin ^= 1;
    }
    // final linear 64 -> 1 (derivatives only)
    float pg = dw3[lane] * zg[wv][pin][lane] + dw3[lane + 32] * zg[wv][pin][lane + 32];
    float ph = dw3[lane] * zh[wv][pin][lane] + dw3[lane + 32] * zh[wv][pin][lane + 32];
#pragma unroll
    for (int off = 16; off > 0; off >>= 1) {
      pg += __shfl_xor(pg, off, 32);
      ph += __shfl_xor(ph, off, 32);
    }
    if (i == 3) dTdt = pg; else lap += ph;
  }
  // q-net with detached interpolated features
  float h = qb1[lane] + c4a[0] * qw1[lane] + c4a[1] * qw1[32 + lane] + c4a[2] * qw1[64 + lane];
#pragma unroll
  for (int c = 0; c < 32; c++) h += ifsh[wv][c] * qw1[(3 + c) * 32 + lane];
  float mu = h;
#pragma unroll
  for (int off = 16; off > 0; off >>= 1) mu += __shfl_xor(mu, off, 32);
  mu *= (1.f / 32.f);
  float dv = h - mu;
  float var = dv * dv;
#pragma unroll
  for (int off = 16; off > 0; off >>= 1) var += __shfl_xor(var, off, 32);
  var *= (1.f / 32.f);
  float hn = tanhf(dv * rsqrtf(var + 1e-5f) * qg[lane] + qbeta[lane]);
  float qp = hn * qw2[lane];
#pragma unroll
  for (int off = 16; off > 0; off >>= 1) qp += __shfl_xor(qp, off, 32);
  float q = tanhf(qp + qb2[0]);
  float alpha = log1pf(expf(alpp[0]));
  if (lane == 0) out[YOFF + point] = dTdt - alpha * lap - q;
}

// ---------------------------------------------------------------------------
extern "C" void kernel_launch(void* const* d_in, const int* in_sizes, int n_in,
                              void* d_out, int out_size, void* d_ws, size_t ws_size,
                              hipStream_t stream) {
  (void)in_sizes; (void)n_in; (void)out_size; (void)ws_size;
  const float* x_data  = (const float*)d_in[0];
  const float* pre_adj = (const float*)d_in[1];
  const float* coords  = (const float*)d_in[2];
  const float* sensors = (const float*)d_in[3];
  const float* adp_adj = (const float*)d_in[4];
  const float* start_w = (const float*)d_in[5];
  const float* start_b = (const float*)d_in[6];
  // layers at 7 + 8*L : fw fb gw gb rw rb sw sb
  const float* e1w = (const float*)d_in[71];
  const float* e1b = (const float*)d_in[72];
  const float* e2w = (const float*)d_in[73];
  const float* e2b = (const float*)d_in[74];
  const float* fpw = (const float*)d_in[75];
  const float* fpb = (const float*)d_in[76];
  const float* dw0 = (const float*)d_in[77];
  const float* dw1 = (const float*)d_in[78];
  const float* dw2 = (const float*)d_in[79];
  const float* dw3 = (const float*)d_in[80];
  const float* db0 = (const float*)d_in[81];
  const float* db1 = (const float*)d_in[82];
  const float* db2 = (const float*)d_in[83];
  const float* db3 = (const float*)d_in[84];
  const float* qw1 = (const float*)d_in[85];
  const float* qb1 = (const float*)d_in[86];
  const float* qgm = (const float*)d_in[87];
  const float* qbt = (const float*)d_in[88];
  const float* qw2 = (const float*)d_in[89];
  const float* qb2 = (const float*)d_in[90];
  const float* sig = (const float*)d_in[91];
  const float* alp = (const float*)d_in[92];

  char* w = (char*)d_ws;
  auto take = [&](size_t bytes) { void* p = (void*)w; w += (bytes + 255) & ~(size_t)255; return p; };
  float*  X0    = (float*)take(134217728);   // (B,64,32,512) f32
  float*  X1    = (float*)take(134217728);
  __bf16* XC    = (__bf16*)take(67108864);   // x_conv bf16
  __bf16* ADJT  = (__bf16*)take(524288);     // adj transposed bf16
  float*  SKIP  = (float*)take(16777216);    // (B,256,512)
  float*  XCL   = (float*)take(2097152);     // x_conv last t slice f32
  float*  FEATS = (float*)take(33554432);    // (B,512,512)
  float*  NF    = (float*)take(2097152);     // node features (B,N,32)

  adj_prep_kernel<<<512, 256, 0, stream>>>(adp_adj, pre_adj, ADJT);
  start_conv_kernel<<<(32u << 20) / 256, 256, 0, stream>>>(x_data, start_w, start_b, X0);

  const int DILSa[8] = {1, 2, 4, 8, 1, 2, 4, 8};
  float* Xcur = X0; float* Xnxt = X1;
  int T = 64;
  for (int L = 0; L < 8; L++) {
    int d = DILSa[L];
    int T_in = T, T_out = T - d;
    const float* fw = (const float*)d_in[7 + 8 * L + 0];
    const float* fb = (const float*)d_in[7 + 8 * L + 1];
    const float* gw = (const float*)d_in[7 + 8 * L + 2];
    const float* gb = (const float*)d_in[7 + 8 * L + 3];
    const float* rw = (const float*)d_in[7 + 8 * L + 4];
    const float* rb = (const float*)d_in[7 + 8 * L + 5];
    const float* sw = (const float*)d_in[7 + 8 * L + 6];
    const float* sb = (const float*)d_in[7 + 8 * L + 7];
    dilconv_kernel<<<32 * T_out * 64, 256, 0, stream>>>(Xcur, fw, fb, gw, gb, XC, XCL, T_in, T_out, d);
    skip_kernel<<<(32 * 256 * 512) / 256, 256, 0, stream>>>(XCL, sw, sb, SKIP, L == 0 ? 1 : 0);
    graphres_kernel<<<dim3(8, T_out, 32), 128, 0, stream>>>(XC, ADJT, rw, rb, Xcur, Xnxt, T_in, T_out, d);
    float* tmp = Xcur; Xcur = Xnxt; Xnxt = tmp;
    T = T_out;
  }
  feats_kernel<<<(32 * 512 * 512) / 256, 256, 0, stream>>>(SKIP, e1w, e1b, FEATS);
  heads_kernel<<<(32 * 512) / 256, 256, 0, stream>>>(FEATS, e2w, e2b, fpw, fpb, (float*)d_out, NF);
  pde_kernel<<<4096, 256, 0, stream>>>(coords, sensors, NF,
      dw0, db0, dw1, db1, dw2, db2, dw3, db3,
      qw1, qb1, qgm, qbt, qw2, qb2, sig, alp, (float*)d_out);
}
